// Mamba2Layer_11476152615397
// MI455X (gfx1250) — matrix operand order repaired and checked by reading, not compiled, not run on hardware
//
#include <hip/hip_runtime.h>

// ---------------------------------------------------------------------------
// Mamba2 forward for MI455X (gfx1250, wave32, WMMA, async-to-LDS).
// Pipeline:
//   0) one-time bf16 conversions: hs (row-major), W_in^T, W_out^T (N x K)
//   1) in_proj GEMM  (bf16 WMMA, f32 accum, LDS-free fragments from global)
//   2) causal depthwise conv (K=4) + SiLU on xBC slice
//   3) sequential SSM scan; x/B/C staged with global_load_async_to_lds_b32
//   4) RMSNorm * gamma * silu(z) -> bf16 directly
//   5) out_proj GEMM (bf16 WMMA, f32 accum)
// ---------------------------------------------------------------------------

#define BATCH      2
#define SEQLEN     2048
#define D_MODEL    1024
#define D_SSM      2048
#define D_STATE    128
#define NHEADS     32
#define HEADDIM    64
#define CONV_DIM   2304          // D_SSM + 2*D_STATE
#define D_IN_PROJ  4384          // D_SSM + CONV_DIM + NHEADS
#define ROWS       (BATCH * SEQLEN)   // 4096

typedef __bf16 bf16x16 __attribute__((ext_vector_type(16)));
typedef float  f32x8   __attribute__((ext_vector_type(8)));

union FragBF {
    bf16x16 v;
    uint4   q[2];
};

__device__ __forceinline__ unsigned short f32_to_bf16_rne(float f) {
    unsigned int u = __float_as_uint(f);
    unsigned int r = u + 0x7FFFu + ((u >> 16) & 1u);
    return (unsigned short)(r >> 16);
}

__device__ __forceinline__ float silu(float x) {
    return x / (1.0f + __expf(-x));
}

// ---------------------------------------------------------------------------
// One-time conversions.
// ---------------------------------------------------------------------------
__global__ __launch_bounds__(256)
void convert_bf16_kernel(const float* __restrict__ src,
                         unsigned short* __restrict__ dst, size_t n) {
    size_t i = (size_t)blockIdx.x * 256 + threadIdx.x;
    if (i < n) dst[i] = f32_to_bf16_rne(src[i]);
}

// src is R x C row-major (R = K, C = N); dst is C x K row-major (B transposed).
__global__ __launch_bounds__(256)
void convert_bf16_transpose_kernel(const float* __restrict__ src,
                                   unsigned short* __restrict__ dst,
                                   int R, int C) {
    size_t i = (size_t)blockIdx.x * 256 + threadIdx.x;
    if (i >= (size_t)R * C) return;
    int r = (int)(i / C);
    int c = (int)(i % C);
    dst[(size_t)c * R + r] = f32_to_bf16_rne(src[i]);
}

// ---------------------------------------------------------------------------
// GEMM  C[M,N] = A[M,K] * B[K,N]; A bf16 row-major, Bt bf16 N x K (B^T),
// C fp32 row-major. LDS-free: WMMA fragments loaded directly from global
// per the ISA 16x32 / 32x16 bf16 VGPR layouts (two b128 loads per fragment).
// Block = 128 threads (4 waves); wave computes a 64x32 tile: 4 A-frags,
// 2 B-frags (4-way reuse), 8 WMMAs per K-step.
// Grid: (N/32, M/256). Requires M % 256 == 0, N % 32 == 0, K % 32 == 0.
// ---------------------------------------------------------------------------
__global__ __launch_bounds__(128)
void gemm_bf16_wmma(const unsigned short* __restrict__ A,
                    const unsigned short* __restrict__ Bt,
                    float* __restrict__ C,
                    int M, int N, int K) {
    const int lane = threadIdx.x & 31;
    const int w    = threadIdx.x >> 5;

    const int n0 = blockIdx.x * 32;
    const int m0 = (blockIdx.y * 4 + w) * 64;

    // A fragment: lane<16 -> M = lane,  K in [0..7] then [16..23]
    //             lane>=16 -> M = lane-16, K ranges shifted by +8
    const int aoff = (lane & 16) ? 8 : 0;
    // B fragment (from B^T): lane<16 -> N = lane, K in [0..7] then [16..23]; lane>=16 -> shifted by +8
    const int bk   = (lane & 16) ? 8 : 0;

    const unsigned short* Arow[4];
    #pragma unroll
    for (int i = 0; i < 4; ++i)
        Arow[i] = A + (size_t)(m0 + 16 * i + (lane & 15)) * K;
    const unsigned short* Bcol0 = Bt + (size_t)(n0 + (lane & 15)) * K;
    const unsigned short* Bcol1 = Bcol0 + (size_t)16 * K;

    f32x8 acc0[4] = {}, acc1[4] = {};

    for (int k0 = 0; k0 < K; k0 += 32) {
        FragBF a[4], b0, b1;
        #pragma unroll
        for (int i = 0; i < 4; ++i) {
            a[i].q[0] = *(const uint4*)(Arow[i] + k0 + aoff);
            a[i].q[1] = *(const uint4*)(Arow[i] + k0 + aoff + 16);
        }
        b0.q[0] = *(const uint4*)(Bcol0 + k0 + bk);
        b0.q[1] = *(const uint4*)(Bcol0 + k0 + bk + 16);
        b1.q[0] = *(const uint4*)(Bcol1 + k0 + bk);
        b1.q[1] = *(const uint4*)(Bcol1 + k0 + bk + 16);

        if (k0 + 64 < K) {   // uniform branch; prefetch 2 K-steps ahead
            #pragma unroll
            for (int i = 0; i < 4; ++i)
                __builtin_prefetch(Arow[i] + k0 + 64, 0, 3);
            __builtin_prefetch(Bcol0 + k0 + 64, 0, 3);
            __builtin_prefetch(Bcol1 + k0 + 64, 0, 3);
        }

        #pragma unroll
        for (int i = 0; i < 4; ++i) {
            acc0[i] = __builtin_amdgcn_wmma_f32_16x16x32_bf16(
                          false, a[i].v, false, b0.v, (short)0, acc0[i], false, false);
            acc1[i] = __builtin_amdgcn_wmma_f32_16x16x32_bf16(
                          false, a[i].v, false, b1.v, (short)0, acc1[i], false, false);
        }
    }

    // D layout: VGPR v -> M = v (+8 for lanes 16..31), N = lane % 16
    const int ncol = lane & 15;
    const int moff = (lane & 16) ? 8 : 0;
    #pragma unroll
    for (int i = 0; i < 4; ++i) {
        #pragma unroll
        for (int v = 0; v < 8; ++v) {
            float* row = C + (size_t)(m0 + 16 * i + moff + v) * N + n0;
            row[ncol]      = acc0[i][v];
            row[16 + ncol] = acc1[i][v];
        }
    }
}

// ---------------------------------------------------------------------------
// Causal depthwise conv (K=4) + SiLU on xBC slice of zxbcdt.
// ---------------------------------------------------------------------------
__global__ __launch_bounds__(256)
void conv_silu_kernel(const float* __restrict__ zxbcdt,
                      const float* __restrict__ conv_w,
                      const float* __restrict__ conv_b,
                      float* __restrict__ xBC) {
    const size_t total = (size_t)BATCH * SEQLEN * CONV_DIM;
    size_t idx = (size_t)blockIdx.x * 256 + threadIdx.x;
    if (idx >= total) return;
    const int c = (int)(idx % CONV_DIM);
    const int l = (int)((idx / CONV_DIM) % SEQLEN);
    const int b = (int)(idx / ((size_t)CONV_DIM * SEQLEN));

    float acc = conv_b[c];
    #pragma unroll
    for (int k = 0; k < 4; ++k) {
        const int ls = l - 3 + k;
        if (ls >= 0) {
            acc += zxbcdt[((size_t)b * SEQLEN + ls) * D_IN_PROJ + D_SSM + c] *
                   conv_w[k * CONV_DIM + c];
        }
    }
    xBC[idx] = silu(acc);
}

// ---------------------------------------------------------------------------
// Sequential SSM scan. One block per (batch, head); 256 threads.
// Thread t owns p = t/4, n-range [(t%4)*32, +32); state lives in registers.
// Per-timestep x/B/C staging uses CDNA5 async copy (global -> LDS) with
// ASYNCcnt completion, instead of load + ds_store round trips.
// BCs[0..127] = B_t, BCs[128..255] = C_t; global source offset D_SSM + t
// covers both (D_SSM+128+(t-128) == D_SSM+t), so the async op is uniform.
// ---------------------------------------------------------------------------
__global__ __launch_bounds__(256)
void mamba_scan_kernel(const float* __restrict__ zxbcdt,
                       const float* __restrict__ xBC,
                       const float* __restrict__ A_log,
                       const float* __restrict__ Dv,
                       const float* __restrict__ dt_bias,
                       float* __restrict__ y_ssm) {
    const int b = blockIdx.x >> 5;   // 0..1
    const int h = blockIdx.x & 31;   // 0..31
    const int t = threadIdx.x;
    const int p  = t >> 2;
    const int nb = (t & 3) * 32;

    __shared__ float xs[HEADDIM];
    __shared__ float BCs[2 * D_STATE];   // [0..127]=B, [128..255]=C
    __shared__ float ypart[256];

    float hstate[32];
    #pragma unroll
    for (int i = 0; i < 32; ++i) hstate[i] = 0.0f;

    const float Ah    = -__expf(A_log[h]);
    const float dbias = dt_bias[h];
    const float Dh    = Dv[h];

    // Loop-invariant LDS byte offsets (low 32 bits of flat shared address).
    const unsigned ldsBC = (unsigned)(size_t)&BCs[t];
    const unsigned ldsX  = (unsigned)(size_t)&xs[t & (HEADDIM - 1)];

    for (int l = 0; l < SEQLEN; ++l) {
        const size_t rowBC = ((size_t)b * SEQLEN + l) * CONV_DIM;
        const size_t rowZ  = ((size_t)b * SEQLEN + l) * D_IN_PROJ;

        // Async stage B_t / C_t (uniform across all 256 lanes).
        {
            const unsigned long long g =
                (unsigned long long)(size_t)(xBC + rowBC + D_SSM + t);
            asm volatile("global_load_async_to_lds_b32 %0, %1, off"
                         :: "v"(ldsBC), "v"(g) : "memory");
        }
        // Async stage x_t (lanes 0..63).
        if (t < HEADDIM) {
            const unsigned long long g =
                (unsigned long long)(size_t)(xBC + rowBC + h * HEADDIM + t);
            asm volatile("global_load_async_to_lds_b32 %0, %1, off"
                         :: "v"(ldsX), "v"(g) : "memory");
        }

        const float dpre  = zxbcdt[rowZ + D_SSM + CONV_DIM + h] + dbias;
        const float delta = (dpre > 20.0f) ? dpre : log1pf(__expf(dpre));
        const float dA    = __expf(delta * Ah);

        asm volatile("s_wait_asynccnt 0" ::: "memory");
        __syncthreads();

        const float xv = xs[p];
        const float dx = delta * xv;
        float acc = 0.0f;
        #pragma unroll
        for (int j = 0; j < 32; ++j) {
            const int n = nb + j;
            hstate[j] = dA * hstate[j] + dx * BCs[n];
            acc += hstate[j] * BCs[D_STATE + n];
        }
        ypart[t] = acc;
        __syncthreads();

        if (t < HEADDIM) {
            const float y = ypart[4 * t] + ypart[4 * t + 1] +
                            ypart[4 * t + 2] + ypart[4 * t + 3] + Dh * xs[t];
            y_ssm[((size_t)b * SEQLEN + l) * D_SSM + h * HEADDIM + t] = y;
        }
        __syncthreads();   // protect shared buffers for next timestep
    }
}

// ---------------------------------------------------------------------------
// RMSNorm over D_SSM, * gamma, * silu(z); reads f32 y_ssm, writes bf16 y
// (feeds GEMM2 directly). One block (256 threads) per row; 8 elems/thread.
// ---------------------------------------------------------------------------
__global__ __launch_bounds__(256)
void rmsnorm_gate_kernel(const float* __restrict__ y_ssm,
                         const float* __restrict__ zxbcdt,
                         const float* __restrict__ gamma,
                         unsigned short* __restrict__ y_bf16) {
    const int row = blockIdx.x;      // 0..ROWS-1
    const int t   = threadIdx.x;
    __shared__ float red[256];

    const size_t base  = (size_t)row * D_SSM;
    const size_t zbase = (size_t)row * D_IN_PROJ;

    float vals[8];
    float ss = 0.0f;
    #pragma unroll
    for (int i = 0; i < 8; ++i) {
        const float v = y_ssm[base + t * 8 + i];
        vals[i] = v;
        ss += v * v;
    }
    red[t] = ss;
    __syncthreads();
    for (int s = 128; s > 0; s >>= 1) {
        if (t < s) red[t] += red[t + s];
        __syncthreads();
    }
    const float inv = rsqrtf(red[0] / (float)D_SSM + 1e-5f);

    #pragma unroll
    for (int i = 0; i < 8; ++i) {
        const int c = t * 8 + i;
        const float z = zxbcdt[zbase + c];
        y_bf16[base + c] = f32_to_bf16_rne(vals[i] * inv * gamma[c] * silu(z));
    }
}

// ---------------------------------------------------------------------------
// Host-side launcher.
// ---------------------------------------------------------------------------
extern "C" void kernel_launch(void* const* d_in, const int* in_sizes, int n_in,
                              void* d_out, int out_size, void* d_ws, size_t ws_size,
                              hipStream_t stream) {
    const float* hs      = (const float*)d_in[0];
    const float* W_in    = (const float*)d_in[1];
    const float* conv_w  = (const float*)d_in[2];
    const float* conv_b  = (const float*)d_in[3];
    const float* A_log   = (const float*)d_in[4];
    const float* Dv      = (const float*)d_in[5];
    const float* dt_bias = (const float*)d_in[6];
    const float* gamma   = (const float*)d_in[7];
    const float* W_out   = (const float*)d_in[8];
    float* out = (float*)d_out;

    // Workspace layout: f32 activations, then bf16 operands (~182 MB total).
    char* p = (char*)d_ws;
    float* zxbcdt = (float*)p;  p += (size_t)ROWS * D_IN_PROJ * 4;     // 71.8 MB
    float* xBC    = (float*)p;  p += (size_t)ROWS * CONV_DIM  * 4;     // 37.7 MB
    float* y_ssm  = (float*)p;  p += (size_t)ROWS * D_SSM     * 4;     // 33.6 MB
    unsigned short* hs_bf   = (unsigned short*)p;  p += (size_t)ROWS * D_MODEL * 2;
    unsigned short* Win_t   = (unsigned short*)p;  p += (size_t)D_IN_PROJ * D_MODEL * 2;
    unsigned short* Wout_t  = (unsigned short*)p;  p += (size_t)D_MODEL * D_SSM * 2;
    unsigned short* y_bf    = (unsigned short*)p;

    // 0) conversions
    {
        const size_t n_hs = (size_t)ROWS * D_MODEL;
        convert_bf16_kernel<<<(unsigned)((n_hs + 255) / 256), 256, 0, stream>>>(
            hs, hs_bf, n_hs);
        const size_t n_wi = (size_t)D_MODEL * D_IN_PROJ;
        convert_bf16_transpose_kernel<<<(unsigned)((n_wi + 255) / 256), 256, 0, stream>>>(
            W_in, Win_t, D_MODEL, D_IN_PROJ);
        const size_t n_wo = (size_t)D_SSM * D_MODEL;
        convert_bf16_transpose_kernel<<<(unsigned)((n_wo + 255) / 256), 256, 0, stream>>>(
            W_out, Wout_t, D_SSM, D_MODEL);
    }
    // 1) in_proj GEMM: [4096 x 1024] * [1024 x 4384]
    {
        dim3 grid(D_IN_PROJ / 32, ROWS / 256);
        gemm_bf16_wmma<<<grid, 128, 0, stream>>>(hs_bf, Win_t, zxbcdt,
                                                 ROWS, D_IN_PROJ, D_MODEL);
    }
    // 2) depthwise conv + SiLU
    {
        const size_t total = (size_t)BATCH * SEQLEN * CONV_DIM;
        conv_silu_kernel<<<(unsigned)((total + 255) / 256), 256, 0, stream>>>(
            zxbcdt, conv_w, conv_b, xBC);
    }
    // 3) SSM scan (async-to-LDS staging)
    mamba_scan_kernel<<<BATCH * NHEADS, 256, 0, stream>>>(
        zxbcdt, xBC, A_log, Dv, dt_bias, y_ssm);
    // 4) RMSNorm * gamma * silu(z) -> bf16
    rmsnorm_gate_kernel<<<ROWS, 256, 0, stream>>>(y_ssm, zxbcdt, gamma, y_bf);
    // 5) out_proj GEMM: [4096 x 2048] * [2048 x 1024]
    {
        dim3 grid(D_MODEL / 32, ROWS / 256);
        gemm_bf16_wmma<<<grid, 128, 0, stream>>>(y_bf, Wout_t, out,
                                                 ROWS, D_MODEL, D_SSM);
    }
}
